// RWKV_Tmix_x070_70944269795570
// MI455X (gfx1250) — matrix operand order repaired and checked by reading, not compiled, not run on hardware
//
#include <hip/hip_runtime.h>
#include <stdint.h>

// ---------------- problem constants ----------------
#define Bq   4
#define Tq   2048
#define Cq   2048
#define Hq   32
#define Nq   64
#define DWq  64
#define DAq  64
#define DVq  32
#define DGq  128
#define BTq  (Bq * Tq)                    // 8192
#define BTCq ((size_t)BTq * (size_t)Cq)   // 16777216
#define EPS_GN 0.00064f

// ---------------- WMMA types (gfx1250, wave32) ----------------
typedef __attribute__((ext_vector_type(16))) __bf16 v16bf;
typedef __attribute__((ext_vector_type(8)))  float  v8f;

union alignas(32) FragBF {
  unsigned short us[16];
  uint4          q4[2];
  v16bf          v;
};

__device__ __forceinline__ unsigned short f2bf(float f) {
  unsigned int u = __float_as_uint(f);
  u += 0x7fffu + ((u >> 16) & 1u);   // round-to-nearest-even
  return (unsigned short)(u >> 16);
}
__device__ __forceinline__ float sigmoidf_(float x) { return 1.0f / (1.0f + __expf(-x)); }

// =====================================================================
// 1) time-shift + 6-way mix, fp32 -> bf16
// =====================================================================
__global__ __launch_bounds__(256) void mix_kernel(
    const float* __restrict__ x,
    const float* __restrict__ mr, const float* __restrict__ mw,
    const float* __restrict__ mk, const float* __restrict__ mv,
    const float* __restrict__ ma, const float* __restrict__ mg,
    unsigned short* __restrict__ oxr, unsigned short* __restrict__ oxw,
    unsigned short* __restrict__ oxk, unsigned short* __restrict__ oxv,
    unsigned short* __restrict__ oxa, unsigned short* __restrict__ oxg)
{
  size_t idx = (size_t)blockIdx.x * blockDim.x + threadIdx.x;
  if (idx >= BTCq) return;
  int    c   = (int)(idx % Cq);
  size_t tok = idx / Cq;
  int    t   = (int)(tok % Tq);
  float xc = x[idx];
  float xp = (t == 0) ? 0.0f : x[idx - Cq];
  float xx = xp - xc;
  oxr[idx] = f2bf(xc + xx * mr[c]);
  oxw[idx] = f2bf(xc + xx * mw[c]);
  oxk[idx] = f2bf(xc + xx * mk[c]);
  oxv[idx] = f2bf(xc + xx * mv[c]);
  oxa[idx] = f2bf(xc + xx * ma[c]);
  oxg[idx] = f2bf(xc + xx * mg[c]);
}

// =====================================================================
// 2) B-operand packer: K-tiled, LDS-image layout.
//    Tile t holds K rows [32t, 32t+32); within a tile, row n occupies 48
//    halves (32 valid + 16 pad) => a GEMM block's B tile is a contiguous
//    128*48 halves = 12288 B region, DMA'd to LDS byte-for-byte.
//    srcMode 0: src is (N x K) row-major  => B(k,n) = src[n*K + k]  (W.T)
//    srcMode 1: src is (K x N) row-major  => B(k,n) = src[k*N + n]  (LoRA)
//    Rows n >= N (up to Nalloc) and pad halves are zero-filled.
// =====================================================================
#define LSTR 48   // halves per packed/LDS row (96 B, 16B-aligned)

__global__ __launch_bounds__(256) void packB_kernel(
    const float* __restrict__ src, unsigned short* __restrict__ dst,
    int K, int N, int Nalloc, int srcMode)
{
  size_t total = (size_t)(K / 32) * Nalloc * LSTR;
  size_t idx = (size_t)blockIdx.x * blockDim.x + threadIdx.x;
  if (idx >= total) return;
  size_t tsz = (size_t)Nalloc * LSTR;
  int t  = (int)(idx / tsz);
  int rem = (int)(idx % tsz);
  int n  = rem / LSTR;
  int kk = rem % LSTR;
  unsigned short v = 0;
  if (kk < 32 && n < N) {
    int k = t * 32 + kk;
    float f = (srcMode == 0) ? src[(size_t)n * K + k] : src[(size_t)k * N + n];
    v = f2bf(f);
  }
  dst[idx] = v;
}

// mode: 0 = tanh, 1 = sigmoid, 2 = copy
__global__ __launch_bounds__(256) void unary_bf16_kernel(
    const float* __restrict__ src, unsigned short* __restrict__ dst, int n, int mode)
{
  int i = blockIdx.x * blockDim.x + threadIdx.x;
  if (i >= n) return;
  float x = src[i];
  if (mode == 0)      x = tanhf(x);
  else if (mode == 1) x = sigmoidf_(x);
  dst[i] = f2bf(x);
}

// =====================================================================
// 3) bf16 WMMA GEMM: C[M,N](f32) = A[M,K](bf16, row-major) * B (packed)
//    Tile 128x128x32, 8 waves x (32x64), double-buffered LDS.
//    Both A and B tiles staged with CDNA5 async DMA
//    (global_load_async_to_lds_b128, ASYNCcnt); compute consumes LDS via
//    ds_load_b128 fragments striped per the 16-bit WMMA operand layout.
//    Requires: M % 128 == 0, K % 32 == 0, Nalloc % 128 == 0.
// =====================================================================
#define GTM 128
#define GTN 128
#define GTK 32

__global__ __launch_bounds__(256) void gemm_bf16_kernel(
    const unsigned short* __restrict__ A,   // M x K row-major bf16
    const unsigned short* __restrict__ Bp,  // packed B (see packB_kernel)
    float* __restrict__ Cm,                 // M x N row-major f32
    int M, int N, int Nalloc, int K)
{
  __shared__ unsigned short lA[2][GTM * LSTR];
  __shared__ unsigned short lB[2][GTN * LSTR];

  const int tid  = threadIdx.x;
  const int lane = tid & 31;
  const int wave = tid >> 5;
  const int m0 = blockIdx.x * GTM;
  const int n0 = blockIdx.y * GTN;
  const int wm = (wave & 3) * 32;   // wave M offset in tile
  const int wn = (wave >> 2) * 64;  // wave N offset in tile

  // A staging: 128x32 halves = 512 x 16B chunks, 2 per thread
  const int achunk0 = tid * 2;
  const int arow0 = achunk0 >> 2;
  const int akc0  = (achunk0 & 3) * 8;
  const int arow1 = (achunk0 + 1) >> 2;
  const int akc1  = ((achunk0 + 1) & 3) * 8;
  // B staging: 128*48 halves = 12288 B = 768 x 16B chunks, 3 per thread
  const size_t btile = (size_t)Nalloc * LSTR;                // halves per K-tile
  const size_t bbase = (size_t)n0 * LSTR;                    // region start (halves)

  v8f acc[2][4];
#pragma unroll
  for (int im = 0; im < 2; ++im)
#pragma unroll
    for (int in = 0; in < 4; ++in)
#pragma unroll
      for (int e = 0; e < 8; ++e) acc[im][in][e] = 0.0f;

  const int lrow = lane & 15;
  const int lhi  = lane >> 4;
  const int ntiles = K / GTK;

  // ---------------- prologue: stage tile 0 into buffer 0 ----------------
  {
    unsigned loff0 = (unsigned)(size_t)&lA[0][arow0 * LSTR + akc0];
    unsigned long long ga0 = (unsigned long long)(A + (size_t)(m0 + arow0) * K + akc0);
    asm volatile("global_load_async_to_lds_b128 %0, %1, off" :: "v"(loff0), "v"(ga0) : "memory");
    unsigned loff1 = (unsigned)(size_t)&lA[0][arow1 * LSTR + akc1];
    unsigned long long ga1 = (unsigned long long)(A + (size_t)(m0 + arow1) * K + akc1);
    asm volatile("global_load_async_to_lds_b128 %0, %1, off" :: "v"(loff1), "v"(ga1) : "memory");
#pragma unroll
    for (int i = 0; i < 3; ++i) {
      int cb = tid * 3 + i;                           // 0..767
      unsigned loff = (unsigned)(size_t)&lB[0][0] + (unsigned)(cb * 16);
      unsigned long long ga =
          (unsigned long long)((const char*)Bp + bbase * 2 + (size_t)cb * 16);
      asm volatile("global_load_async_to_lds_b128 %0, %1, off" :: "v"(loff), "v"(ga) : "memory");
    }
    asm volatile("s_wait_asynccnt 0" ::: "memory");
    __syncthreads();
  }

  // ---------------- main loop, double-buffered ----------------
  for (int it = 0; it < ntiles; ++it) {
    const int cur = it & 1;
    const int nxt = cur ^ 1;
    const bool hasNext = (it + 1 < ntiles);
    const int ktn = (it + 1) * GTK;

    if (hasNext) {
      if (it + 2 < ntiles)   // hint the tile after next into L2
        __builtin_prefetch(A + (size_t)(m0 + (tid >> 1)) * K + ktn + GTK, 0, 1);
      // A tile it+1
      unsigned loff0 = (unsigned)(size_t)&lA[nxt][arow0 * LSTR + akc0];
      unsigned long long ga0 = (unsigned long long)(A + (size_t)(m0 + arow0) * K + ktn + akc0);
      asm volatile("global_load_async_to_lds_b128 %0, %1, off" :: "v"(loff0), "v"(ga0) : "memory");
      unsigned loff1 = (unsigned)(size_t)&lA[nxt][arow1 * LSTR + akc1];
      unsigned long long ga1 = (unsigned long long)(A + (size_t)(m0 + arow1) * K + ktn + akc1);
      asm volatile("global_load_async_to_lds_b128 %0, %1, off" :: "v"(loff1), "v"(ga1) : "memory");
      // B tile it+1 (already in LDS image layout)
#pragma unroll
      for (int i = 0; i < 3; ++i) {
        int cb = tid * 3 + i;
        unsigned loff = (unsigned)(size_t)&lB[nxt][0] + (unsigned)(cb * 16);
        unsigned long long ga = (unsigned long long)
            ((const char*)Bp + ((size_t)(it + 1) * btile + bbase) * 2 + (size_t)cb * 16);
        asm volatile("global_load_async_to_lds_b128 %0, %1, off" :: "v"(loff), "v"(ga) : "memory");
      }
    }

    // ---- compute tile `it` from buffer `cur` ----
    FragBF fa[2], fb[4];
#pragma unroll
    for (int im = 0; im < 2; ++im) {
      const unsigned short* p = &lA[cur][(wm + im * 16 + lrow) * LSTR + 8 * lhi];
      fa[im].q4[0] = *(const uint4*)p;          // K = 8*hi .. 8*hi+7
      fa[im].q4[1] = *(const uint4*)(p + 16);   // K = 16+8*hi .. 23+8*hi
    }
#pragma unroll
    for (int in = 0; in < 4; ++in) {
      const unsigned short* p = &lB[cur][(wn + in * 16 + lrow) * LSTR + 8 * lhi];
      fb[in].q4[0] = *(const uint4*)p;          // K = 8*hi .. 8*hi+7
      fb[in].q4[1] = *(const uint4*)(p + 16);   // K = 16+8*hi .. 23+8*hi
    }
#pragma unroll
    for (int im = 0; im < 2; ++im)
#pragma unroll
      for (int in = 0; in < 4; ++in)
        acc[im][in] = __builtin_amdgcn_wmma_f32_16x16x32_bf16(
            false, fa[im].v, false, fb[in].v, (short)0, acc[im][in], false, false);

    asm volatile("s_wait_asynccnt 0" ::: "memory");
    __syncthreads();
  }

  // ---- store C: lane gives col = lane&15, VGPR e gives row = e + 8*hi ----
#pragma unroll
  for (int im = 0; im < 2; ++im)
#pragma unroll
    for (int in = 0; in < 4; ++in) {
      int col = n0 + wn + in * 16 + lrow;
      if (col < N) {
        size_t rowbase = (size_t)(m0 + wm + im * 16 + 8 * lhi);
#pragma unroll
        for (int e = 0; e < 8; ++e)
          Cm[(rowbase + e) * N + col] = acc[im][in][e];
      }
    }
}

// =====================================================================
// 4) fused post-GEMM: decay w, sigmoid a / v-gate mix, kk L2-norm per
//    head, k final. One workgroup per token (256 thr x 8 ch).
// =====================================================================
__global__ __launch_bounds__(256) void post_kernel(
    float* __restrict__ kf, float* __restrict__ dw, float* __restrict__ da,
    const float* __restrict__ dvg, float* __restrict__ vv,
    const float* __restrict__ vfirst, float* __restrict__ kkout,
    const float* __restrict__ w0, const float* __restrict__ a0,
    const float* __restrict__ v0, const float* __restrict__ k_k,
    const float* __restrict__ k_a)
{
  __shared__ float sred[256];
  __shared__ float hinv[32];
  int tid = threadIdx.x;
  size_t tokbase = (size_t)blockIdx.x * Cq;
  int c0 = tid * 8;

  float kv[8], kkpre[8];
  float ss = 0.0f;
#pragma unroll
  for (int e = 0; e < 8; ++e) {
    int c = c0 + e;
    float kr = kf[tokbase + c];
    kv[e] = kr;
    float kp = kr * k_k[c];
    kkpre[e] = kp;
    ss += kp * kp;
  }
  sred[tid] = ss;
  __syncthreads();
  if ((tid & 7) == 0) {
    float s = 0.0f;
#pragma unroll
    for (int e = 0; e < 8; ++e) s += sred[tid + e];
    hinv[tid >> 3] = 1.0f / fmaxf(sqrtf(s), 1e-12f);
  }
  __syncthreads();
  float inv = hinv[tid >> 3];

#pragma unroll
  for (int e = 0; e < 8; ++e) {
    int c = c0 + e;
    size_t gdx = tokbase + c;
    kkout[gdx] = kkpre[e] * inv;
    float av = sigmoidf_(a0[c] + da[gdx]);
    da[gdx] = av;
    kf[gdx] = kv[e] * (1.0f + (av - 1.0f) * k_a[c]);
    float z  = w0[c] + dw[gdx];               // w = -softplus(-z) - 0.5
    float y  = -z;
    float sp = fmaxf(y, 0.0f) + log1pf(__expf(-fabsf(y)));
    float w  = -sp - 0.5f;
    dw[gdx]  = __expf(-__expf(w));            // store decay directly
    float gate = sigmoidf_(v0[c] + dvg[gdx]);
    float vr = vv[gdx];
    vv[gdx] = vr + (vfirst[gdx] - vr) * gate;
  }
}

// =====================================================================
// 5) sequential per-head matrix-state scan. 128 blocks = (b,h); 256 thr:
//    thread = (row i in 0..63) x (quarter q in 0..3 over 16 j columns).
// =====================================================================
struct alignas(16) V16 { float f[16]; };

__global__ __launch_bounds__(256) void scan_kernel(
    const float* __restrict__ r, const float* __restrict__ wdec,
    const float* __restrict__ k, const float* __restrict__ kk,
    const float* __restrict__ a, const float* __restrict__ v,
    float* __restrict__ o)
{
  int bh = blockIdx.x;
  int b = bh / Hq, h = bh % Hq;
  int tid = threadIdx.x;
  int i = tid >> 2, q = tid & 3, j0 = q << 4;

  float st[16];
#pragma unroll
  for (int jj = 0; jj < 16; ++jj) st[jj] = 0.0f;

  size_t base = (size_t)b * Tq * Cq + (size_t)h * Nq;
  for (int t = 0; t < Tq; ++t, base += Cq) {
    V16 wv  = *(const V16*)(wdec + base + j0);
    V16 kkv = *(const V16*)(kk   + base + j0);
    V16 av  = *(const V16*)(a    + base + j0);
    V16 kv  = *(const V16*)(k    + base + j0);
    V16 rv  = *(const V16*)(r    + base + j0);
    float vi = v[base + i];

    float sa = 0.0f;
#pragma unroll
    for (int jj = 0; jj < 16; ++jj) sa -= st[jj] * kkv.f[jj];   // state @ (-kk)
    sa += __shfl_xor(sa, 1, 32);
    sa += __shfl_xor(sa, 2, 32);

    float outp = 0.0f;
#pragma unroll
    for (int jj = 0; jj < 16; ++jj) {
      float bj = kkv.f[jj] * av.f[jj];                          // b = kk*a
      st[jj] = st[jj] * wv.f[jj] + sa * bj + vi * kv.f[jj];
      outp += st[jj] * rv.f[jj];
    }
    outp += __shfl_xor(outp, 1, 32);
    outp += __shfl_xor(outp, 2, 32);
    if (q == 0) o[base + i] = outp;
  }
}

// =====================================================================
// 6) fused GroupNorm (per head) + rk bonus + gate, -> bf16 for final GEMM
// =====================================================================
__global__ __launch_bounds__(256) void gn_kernel(
    const float* __restrict__ o, const float* __restrict__ r,
    const float* __restrict__ kf, const float* __restrict__ v,
    const float* __restrict__ g, const float* __restrict__ r_k,
    const float* __restrict__ ln_w, const float* __restrict__ ln_b,
    unsigned short* __restrict__ ygate)
{
  __shared__ float s0[256], s1[256], s2[256];
  __shared__ float hm[32], hvv[32], hb[32];
  int tid = threadIdx.x;
  size_t tokbase = (size_t)blockIdx.x * Cq;
  int c0 = tid * 8;
  int h = tid >> 3;

  float ov[8];
  float sum = 0.0f, sq = 0.0f, bon = 0.0f;
#pragma unroll
  for (int e = 0; e < 8; ++e) {
    int c = c0 + e;
    size_t gdx = tokbase + c;
    float xo = o[gdx];
    ov[e] = xo;
    sum += xo;
    sq  += xo * xo;
    bon += r[gdx] * kf[gdx] * r_k[h * Nq + (c & (Nq - 1))];
  }
  s0[tid] = sum; s1[tid] = sq; s2[tid] = bon;
  __syncthreads();
  if ((tid & 7) == 0) {
    float A = 0, Bv = 0, Cb = 0;
#pragma unroll
    for (int e = 0; e < 8; ++e) { A += s0[tid + e]; Bv += s1[tid + e]; Cb += s2[tid + e]; }
    float mu = A / (float)Nq;
    hm[h]  = mu;
    hvv[h] = Bv / (float)Nq - mu * mu;
    hb[h]  = Cb;
  }
  __syncthreads();
  float mu = hm[h];
  float inv = rsqrtf(hvv[h] + EPS_GN);
  float bonus = hb[h];
#pragma unroll
  for (int e = 0; e < 8; ++e) {
    int c = c0 + e;
    size_t gdx = tokbase + c;
    float y = (ov[e] - mu) * inv * ln_w[c] + ln_b[c] + bonus * v[gdx];
    ygate[gdx] = f2bf(y * g[gdx]);
  }
}

// =====================================================================
// launcher
// =====================================================================
extern "C" void kernel_launch(void* const* d_in, const int* in_sizes, int n_in,
                              void* d_out, int out_size, void* d_ws, size_t ws_size,
                              hipStream_t stream) {
  (void)in_sizes; (void)n_in; (void)out_size; (void)ws_size;

  const float* x       = (const float*)d_in[0];
  const float* v_first = (const float*)d_in[1];
  const float* x_r = (const float*)d_in[2];
  const float* x_w = (const float*)d_in[3];
  const float* x_k = (const float*)d_in[4];
  const float* x_v = (const float*)d_in[5];
  const float* x_a = (const float*)d_in[6];
  const float* x_g = (const float*)d_in[7];
  const float* w0  = (const float*)d_in[8];
  const float* w1  = (const float*)d_in[9];
  const float* w2  = (const float*)d_in[10];
  const float* a0  = (const float*)d_in[11];
  const float* a1  = (const float*)d_in[12];
  const float* a2  = (const float*)d_in[13];
  const float* v0  = (const float*)d_in[14];
  const float* v1  = (const float*)d_in[15];
  const float* v2  = (const float*)d_in[16];
  const float* g1  = (const float*)d_in[17];
  const float* g2  = (const float*)d_in[18];
  const float* k_k = (const float*)d_in[19];
  const float* k_a = (const float*)d_in[20];
  const float* r_k = (const float*)d_in[21];
  const float* Wr  = (const float*)d_in[22];
  const float* Wk  = (const float*)d_in[23];
  const float* Wv  = (const float*)d_in[24];
  const float* Wo  = (const float*)d_in[25];
  const float* ln_w = (const float*)d_in[26];
  const float* ln_b = (const float*)d_in[27];

  char* wsp = (char*)d_ws;
  size_t off = 0;
  auto alloc = [&](size_t bytes) -> void* {
    void* p = wsp + off;
    off += (bytes + 255) & ~(size_t)255;
    return p;
  };
  // packed-B size in bytes
  auto packBytes = [](int K, int Nalloc) -> size_t {
    return (size_t)(K / 32) * Nalloc * LSTR * 2;
  };

  // bf16 activation buffers
  unsigned short* xr_bf = (unsigned short*)alloc(BTCq * 2);
  unsigned short* xw_bf = (unsigned short*)alloc(BTCq * 2);
  unsigned short* xk_bf = (unsigned short*)alloc(BTCq * 2);
  unsigned short* xv_bf = (unsigned short*)alloc(BTCq * 2);
  unsigned short* xa_bf = (unsigned short*)alloc(BTCq * 2);
  unsigned short* xg_bf = (unsigned short*)alloc(BTCq * 2);
  // packed bf16 B operands
  unsigned short* WrP = (unsigned short*)alloc(packBytes(Cq, Cq));
  unsigned short* WkP = (unsigned short*)alloc(packBytes(Cq, Cq));
  unsigned short* WvP = (unsigned short*)alloc(packBytes(Cq, Cq));
  unsigned short* WoP = (unsigned short*)alloc(packBytes(Cq, Cq));
  unsigned short* w1P = (unsigned short*)alloc(packBytes(Cq, 128));
  unsigned short* a1P = (unsigned short*)alloc(packBytes(Cq, 128));
  unsigned short* v1P = (unsigned short*)alloc(packBytes(Cq, 128));
  unsigned short* g1P = (unsigned short*)alloc(packBytes(Cq, 128));
  unsigned short* w2P = (unsigned short*)alloc(packBytes(DWq, Cq));
  unsigned short* a2P = (unsigned short*)alloc(packBytes(DAq, Cq));
  unsigned short* v2P = (unsigned short*)alloc(packBytes(DVq, Cq));
  unsigned short* g2P = (unsigned short*)alloc(packBytes(DGq, Cq));
  // fp32 intermediates (several reused in place)
  float* r_f   = (float*)alloc(BTCq * 4);
  float* k_f   = (float*)alloc(BTCq * 4);   // kraw -> kfin
  float* v_f   = (float*)alloc(BTCq * 4);   // vraw -> v
  float* dw_f  = (float*)alloc(BTCq * 4);   // dw -> exp(-exp(w))
  float* da_f  = (float*)alloc(BTCq * 4);   // da -> a
  float* dvg_f = (float*)alloc(BTCq * 4);   // v-gate logits, then reused as o
  float* kk_f  = (float*)alloc(BTCq * 4);
  float* g_f   = (float*)alloc(BTCq * 4);
  float* hw_f  = (float*)alloc((size_t)BTq * DWq * 4);
  float* ha_f  = (float*)alloc((size_t)BTq * DAq * 4);
  float* hv_f  = (float*)alloc((size_t)BTq * DVq * 4);
  float* hg_f  = (float*)alloc((size_t)BTq * DGq * 4);
  unsigned short* hwb = (unsigned short*)alloc((size_t)BTq * DWq * 2);
  unsigned short* hab = (unsigned short*)alloc((size_t)BTq * DAq * 2);
  unsigned short* hvb = (unsigned short*)alloc((size_t)BTq * DVq * 2);
  unsigned short* hgb = (unsigned short*)alloc((size_t)BTq * DGq * 2);
  float* o_f = dvg_f;                        // dvg dead after post_kernel
  unsigned short* ygate = xr_bf;             // xr dead after r GEMM

  const int EW = 256;
  // 1) mixes
  mix_kernel<<<(unsigned)((BTCq + EW - 1) / EW), EW, 0, stream>>>(
      x, x_r, x_w, x_k, x_v, x_a, x_g, xr_bf, xw_bf, xk_bf, xv_bf, xa_bf, xg_bf);
  // 2) pack every B operand into the LDS-image layout
  auto packGrid = [&](int K, int Nalloc) -> unsigned {
    return (unsigned)((packBytes(K, Nalloc) / 2 + EW - 1) / EW);
  };
  packB_kernel<<<packGrid(Cq, Cq), EW, 0, stream>>>(Wr, WrP, Cq, Cq, Cq, 0);
  packB_kernel<<<packGrid(Cq, Cq), EW, 0, stream>>>(Wk, WkP, Cq, Cq, Cq, 0);
  packB_kernel<<<packGrid(Cq, Cq), EW, 0, stream>>>(Wv, WvP, Cq, Cq, Cq, 0);
  packB_kernel<<<packGrid(Cq, Cq), EW, 0, stream>>>(Wo, WoP, Cq, Cq, Cq, 0);
  packB_kernel<<<packGrid(Cq, 128), EW, 0, stream>>>(w1, w1P, Cq, DWq, 128, 1);
  packB_kernel<<<packGrid(Cq, 128), EW, 0, stream>>>(a1, a1P, Cq, DAq, 128, 1);
  packB_kernel<<<packGrid(Cq, 128), EW, 0, stream>>>(v1, v1P, Cq, DVq, 128, 1);
  packB_kernel<<<packGrid(Cq, 128), EW, 0, stream>>>(g1, g1P, Cq, DGq, 128, 1);
  packB_kernel<<<packGrid(DWq, Cq), EW, 0, stream>>>(w2, w2P, DWq, Cq, Cq, 1);
  packB_kernel<<<packGrid(DAq, Cq), EW, 0, stream>>>(a2, a2P, DAq, Cq, Cq, 1);
  packB_kernel<<<packGrid(DVq, Cq), EW, 0, stream>>>(v2, v2P, DVq, Cq, Cq, 1);
  packB_kernel<<<packGrid(DGq, Cq), EW, 0, stream>>>(g2, g2P, DGq, Cq, Cq, 1);

  // 3) big projections + LoRA first stage (WMMA)
  dim3 blk(256);
  dim3 gBig(BTq / GTM, Cq / GTN);
  dim3 gSm(BTq / GTM, 1);
  gemm_bf16_kernel<<<gBig, blk, 0, stream>>>(xr_bf, WrP, r_f, BTq, Cq, Cq, Cq);
  gemm_bf16_kernel<<<gBig, blk, 0, stream>>>(xk_bf, WkP, k_f, BTq, Cq, Cq, Cq);
  gemm_bf16_kernel<<<gBig, blk, 0, stream>>>(xv_bf, WvP, v_f, BTq, Cq, Cq, Cq);
  gemm_bf16_kernel<<<gSm, blk, 0, stream>>>(xw_bf, w1P, hw_f, BTq, DWq, 128, Cq);
  gemm_bf16_kernel<<<gSm, blk, 0, stream>>>(xa_bf, a1P, ha_f, BTq, DAq, 128, Cq);
  gemm_bf16_kernel<<<gSm, blk, 0, stream>>>(xv_bf, v1P, hv_f, BTq, DVq, 128, Cq);
  gemm_bf16_kernel<<<gSm, blk, 0, stream>>>(xg_bf, g1P, hg_f, BTq, DGq, 128, Cq);

  // LoRA activations -> bf16
  unary_bf16_kernel<<<(BTq * DWq + EW - 1) / EW, EW, 0, stream>>>(hw_f, hwb, BTq * DWq, 0);
  unary_bf16_kernel<<<(BTq * DAq + EW - 1) / EW, EW, 0, stream>>>(ha_f, hab, BTq * DAq, 2);
  unary_bf16_kernel<<<(BTq * DVq + EW - 1) / EW, EW, 0, stream>>>(hv_f, hvb, BTq * DVq, 2);
  unary_bf16_kernel<<<(BTq * DGq + EW - 1) / EW, EW, 0, stream>>>(hg_f, hgb, BTq * DGq, 1);

  // LoRA second stage (WMMA)
  gemm_bf16_kernel<<<gBig, blk, 0, stream>>>(hwb, w2P, dw_f, BTq, Cq, Cq, DWq);
  gemm_bf16_kernel<<<gBig, blk, 0, stream>>>(hab, a2P, da_f, BTq, Cq, Cq, DAq);
  gemm_bf16_kernel<<<gBig, blk, 0, stream>>>(hvb, v2P, dvg_f, BTq, Cq, Cq, DVq);
  gemm_bf16_kernel<<<gBig, blk, 0, stream>>>(hgb, g2P, g_f, BTq, Cq, Cq, DGq);

  // 4) fused post-processing (per token)
  post_kernel<<<BTq, 256, 0, stream>>>(k_f, dw_f, da_f, dvg_f, v_f, v_first,
                                       kk_f, w0, a0, v0, k_k, k_a);

  // 5) sequential scan over T, one block per (b,h)
  scan_kernel<<<Bq * Hq, 256, 0, stream>>>(r_f, dw_f, k_f, kk_f, da_f, v_f, o_f);

  // 6) GroupNorm + bonus + gate -> bf16
  gn_kernel<<<BTq, 256, 0, stream>>>(o_f, r_f, k_f, v_f, g_f, r_k, ln_w, ln_b, ygate);

  // 7) output projection straight into d_out (WMMA)
  gemm_bf16_kernel<<<gBig, blk, 0, stream>>>(ygate, WoP, (float*)d_out, BTq, Cq, Cq, Cq);

  // second tuple element: pass-through v_first
  (void)hipMemcpyAsync((float*)d_out + BTCq, v_first, BTCq * sizeof(float),
                       hipMemcpyDeviceToDevice, stream);
}
